// CRFModel_23837068493389
// MI455X (gfx1250) — matrix operand order repaired and checked by reading, not compiled, not run on hardware
//
#include <hip/hip_runtime.h>
#include <hip/hip_bf16.h>

typedef __attribute__((ext_vector_type(16))) _Float16 v16h;
typedef __attribute__((ext_vector_type(8)))  _Float16 v8h_t;
typedef __attribute__((ext_vector_type(8)))  float    v8f;
typedef __attribute__((ext_vector_type(4)))  float    v4f;
typedef unsigned int u32x4 __attribute__((ext_vector_type(4)));
typedef int          i32x4 __attribute__((ext_vector_type(4)));
typedef int          i32x8 __attribute__((ext_vector_type(8)));

#define BB 64
#define TT 2048
#define HH 256
#define UU 32

#if defined(__has_builtin)
#if __has_builtin(__builtin_amdgcn_tensor_load_to_lds) && __has_builtin(__builtin_amdgcn_s_wait_tensorcnt)
#define HAVE_TDM 1
#endif
#endif

#ifdef HAVE_TDM
// 1-D contiguous copy global -> LDS via the Tensor Data Mover.
// n8 = number of 8-byte units. Descriptor per CDNA5 ISA §8 (group0/group1).
__device__ __forceinline__ void tdm_load_1d(const void* gsrc, void* ldst, unsigned n8) {
  const unsigned lds_addr = (unsigned)(unsigned long long)(uintptr_t)ldst; // LDS aperture: low 32 bits = offset
  const unsigned long long ga = (unsigned long long)(uintptr_t)gsrc;
  u32x4 g0 = {};
  g0[0] = 1u;                                   // count=1 (valid user descriptor)
  g0[1] = lds_addr;                             // lds_addr [63:32]
  g0[2] = (unsigned)ga;                         // global_addr[31:0]
  g0[3] = (unsigned)(ga >> 32) | (2u << 30);    // global_addr[56:32] | type=2 ("image")
  i32x8 g1 = {};
  g1[0] = (int)(3u << 16);                      // data_size=3 (8B units)
  g1[1] = (int)((n8 & 0xFFFFu) << 16);          // tensor_dim0[15:0] @ bits 63:48
  g1[2] = (int)((n8 >> 16) | (1u << 16));       // tensor_dim0[31:16] | tensor_dim1=1
  g1[3] = (int)((n8 & 0xFFFFu) << 16);          // tile_dim0 @ bits 127:112
  g1[4] = 0;                                    // tile_dim1=0, tile_dim2=0 (1-D)
  g1[5] = (int)n8;                              // tensor_dim0_stride[31:0]
  g1[6] = 0;
  g1[7] = 0;
  i32x4 g2 = {};
  i32x4 g3 = {};
#if __clang_major__ >= 23
  i32x8 g4 = {};
  __builtin_amdgcn_tensor_load_to_lds(g0, g1, g2, g3, g4, 0);
#else
  __builtin_amdgcn_tensor_load_to_lds(g0, g1, g2, g3, 0);
#endif
}
#endif

// ---------------------------------------------------------------------------
// Kernel 1: repack base_W [H,H] and crf_kernel [H,U] (fp32 row-major) into
// f16 WMMA B-fragment layout (lane l: n = l&15, halves j -> K = 16*(l>>4)+j;
// fragment (chunk c, tile nt) contiguous: idx = ((c*NT+nt)*32 + l)*16 + j).
// ---------------------------------------------------------------------------
__global__ void pack_weights(const float* __restrict__ W,
                             const float* __restrict__ K,
                             _Float16* __restrict__ Wp,
                             _Float16* __restrict__ Kp) {
  int idx = blockIdx.x * blockDim.x + threadIdx.x;
  if (idx < 8 * 16 * 32 * 16) {
    int j  = idx & 15;
    int l  = (idx >> 4) & 31;
    int nt = (idx >> 9) & 15;
    int c  = idx >> 13;
    int n  = nt * 16 + (l & 15);
    int k  = c * 32 + 16 * (l >> 4) + j;
    Wp[idx] = (_Float16)W[k * HH + n];
  } else if (idx < 8 * 16 * 32 * 16 + 8 * 2 * 32 * 16) {
    int r  = idx - 8 * 16 * 32 * 16;
    int j  = r & 15;
    int l  = (r >> 4) & 31;
    int nt = (r >> 9) & 1;
    int c  = r >> 10;
    int n  = nt * 16 + (l & 15);
    int k  = c * 32 + 16 * (l >> 4) + j;
    Kp[r] = (_Float16)K[k * UU + n];
  }
}

__global__ void zero_loss(float* lossp) { *lossp = 0.0f; }

// ---------------------------------------------------------------------------
// Kernel 2: pot = tanh(x@W + b) @ K + crf_bias (+ boundaries).
// 128 threads = 4 waves, 64 rows/block. The packed W is streamed chunk-by-
// chunk into a DOUBLE-BUFFERED LDS stage by the Tensor Data Mover: while the
// waves run WMMAs against chunk c, the TDM is already fetching chunk c+1.
// ---------------------------------------------------------------------------
__global__ __launch_bounds__(128)
void pot_gemm(const float* __restrict__ x, const float* __restrict__ base_b,
              const _Float16* __restrict__ Wp, const _Float16* __restrict__ Kp,
              const float* __restrict__ crf_bias,
              const float* __restrict__ lb, const float* __restrict__ rb,
              float* __restrict__ pot) {
  __shared__ _Float16 hlds[4 * 16 * HH];        // 32 KB: per-wave 16x256 f16 h-tile
  __shared__ _Float16 wlds[2 * 16 * 32 * 16];   // 2 x 16 KB: double-buffered W chunk

  const int lane = threadIdx.x & 31;
  const int w    = threadIdx.x >> 5;
  const int g    = lane >> 4;
  const int ln   = lane & 15;
  const size_t row0 = (size_t)blockIdx.x * 64 + (size_t)w * 16;

  const float* xrow = x + (row0 + (size_t)ln) * HH;

#ifdef HAVE_TDM
  if (threadIdx.x < 32)                         // wave 0: prime the pipeline
    tdm_load_1d(Wp, wlds, 2048u);
#endif

  v8f acc[16] = {};
  for (int c = 0; c < 8; ++c) {
#ifdef HAVE_TDM
    if (threadIdx.x < 32)
      __builtin_amdgcn_s_wait_tensorcnt(0);     // chunk c landed in LDS
    __syncthreads();                            // visible to all 4 waves
    if (c < 7 && threadIdx.x < 32)              // overlap: DMA chunk c+1 now
      tdm_load_1d(Wp + (size_t)(c + 1) * 8192,
                  wlds + (size_t)((c + 1) & 1) * 8192, 2048u);
#else
    __syncthreads();                            // previous chunk consumed
    {
      const v8h_t* src = (const v8h_t*)(Wp + (size_t)c * 8192);
      v8h_t* dst = (v8h_t*)(wlds + (size_t)(c & 1) * 8192);
      for (int i = threadIdx.x; i < 1024; i += 128) dst[i] = src[i];
    }
    __syncthreads();
#endif

    // ---- A fragment: x rows, fp32 -> f16 ----
    const int k0 = c * 32;
    const float* ap = xrow + k0 + 16 * g;
    if (c < 7) __builtin_prefetch(xrow + k0 + 32, 0, 3);
    v4f a0 = *(const v4f*)(ap);
    v4f a1 = *(const v4f*)(ap + 4);
    v4f a2 = *(const v4f*)(ap + 8);
    v4f a3 = *(const v4f*)(ap + 12);
    v16h af;
#pragma unroll
    for (int i = 0; i < 4; ++i) {
      af[i]      = (_Float16)a0[i];
      af[4 + i]  = (_Float16)a1[i];
      af[8 + i]  = (_Float16)a2[i];
      af[12 + i] = (_Float16)a3[i];
    }
    const v16h* wpp = (const v16h*)(wlds + (size_t)(c & 1) * 8192) + lane;
#pragma unroll
    for (int nt = 0; nt < 16; ++nt) {
      v16h bf = wpp[nt * 32];                   // 2x ds_load_b128
      acc[nt] = __builtin_amdgcn_wmma_f32_16x16x32_f16(
          false, af, false, bf, (short)0, acc[nt], false, false);
    }
  }

  // tanh(acc + b) -> f16 LDS, row-major [16][256] per wave
#pragma unroll
  for (int nt = 0; nt < 16; ++nt) {
    const int n = nt * 16 + ln;
    const float bb = base_b[n];
#pragma unroll
    for (int r = 0; r < 8; ++r) {
      const int mm = r + 8 * g;
      hlds[(w * 16 + mm) * HH + n] = (_Float16)tanhf(acc[nt][r] + bb);
    }
  }
  __syncthreads();

  // GEMM2: pot_tile[16x32] = h_f16[16x256] @ Kp
  v8f pacc[2] = {};
  for (int c = 0; c < 8; ++c) {
    const _Float16* hp = &hlds[(w * 16 + ln) * HH + c * 32 + 16 * g];
    v8h_t lo = *(const v8h_t*)hp;
    v8h_t hi = *(const v8h_t*)(hp + 8);
    v16h af;
#pragma unroll
    for (int i = 0; i < 8; ++i) { af[i] = lo[i]; af[8 + i] = hi[i]; }
    const v16h* kpp = (const v16h*)Kp + (size_t)c * 64 + lane;
    pacc[0] = __builtin_amdgcn_wmma_f32_16x16x32_f16(
        false, af, false, kpp[0], (short)0, pacc[0], false, false);
    pacc[1] = __builtin_amdgcn_wmma_f32_16x16x32_f16(
        false, af, false, kpp[32], (short)0, pacc[1], false, false);
  }

#pragma unroll
  for (int ut = 0; ut < 2; ++ut) {
    const int u = ut * 16 + ln;
    const float cb = crf_bias[u], lbv = lb[u], rbv = rb[u];
#pragma unroll
    for (int r = 0; r < 8; ++r) {
      const size_t row = row0 + (size_t)(r + 8 * g);
      const int t = (int)(row & (TT - 1));
      float p = pacc[ut][r] + cb;
      if (t == 0)      p += lbv;
      if (t == TT - 1) p += rbv;
      pot[row * UU + u] = p;
    }
  }
}

// ---------------------------------------------------------------------------
// Kernel 3: fused Viterbi forward + log-norm forward + log-likelihood.
// One wave (32 lanes = 32 tags) per batch element.
// ---------------------------------------------------------------------------
__global__ __launch_bounds__(32)
void viterbi_fwd(const float* __restrict__ pot, const float* __restrict__ chain,
                 const int* __restrict__ y, unsigned char* __restrict__ bp,
                 unsigned char* __restrict__ tags, float* __restrict__ lossp) {
  const int b = blockIdx.x;
  const int j = threadIdx.x;

  float tcol[32];
#pragma unroll
  for (int i = 0; i < 32; ++i) tcol[i] = chain[i * UU + j];

  const float* pb = pot + (size_t)b * TT * UU;
  float alpha    = pb[j];
  float alpha_ln = alpha;

  for (int t = 1; t < TT; ++t) {
    const float potv = pb[t * UU + j];
    float bestv = -INFINITY; int besti = 0;
    float m = -INFINITY, ssum = 0.0f;
#pragma unroll
    for (int i = 0; i < 32; ++i) {
      const float ai  = __shfl(alpha, i, 32);
      const float ali = __shfl(alpha_ln, i, 32);
      const float s = ai + tcol[i];
      if (s > bestv) { bestv = s; besti = i; }
      const float sl = ali + tcol[i];
      const float nm = fmaxf(m, sl);
      ssum = ssum * __expf(m - nm) + __expf(sl - nm);
      m = nm;
    }
    alpha    = bestv + potv;
    alpha_ln = m + __logf(ssum) + potv;
    bp[((size_t)(t - 1) * BB + b) * UU + j] = (unsigned char)besti;
  }

  float bv = alpha; int bi = j;
#pragma unroll
  for (int o = 16; o > 0; o >>= 1) {
    const float ov = __shfl_xor(bv, o, 32);
    const int   oi = __shfl_xor(bi, o, 32);
    if (ov > bv || (ov == bv && oi < bi)) { bv = ov; bi = oi; }
  }
  float mm = alpha_ln;
#pragma unroll
  for (int o = 16; o > 0; o >>= 1) mm = fmaxf(mm, __shfl_xor(mm, o, 32));
  float ex = __expf(alpha_ln - mm);
#pragma unroll
  for (int o = 16; o > 0; o >>= 1) ex += __shfl_xor(ex, o, 32);
  const float logZ = mm + __logf(ex);

  const int* yb = y + (size_t)b * TT;
  float un = 0.0f, bin = 0.0f;
  for (int tt = j; tt < TT; tt += 32) {
    const int yt = yb[tt];
    un += pb[tt * UU + yt];
    if (tt + 1 < TT) bin += chain[yt * UU + yb[tt + 1]];
  }
#pragma unroll
  for (int o = 16; o > 0; o >>= 1) {
    un  += __shfl_xor(un, o, 32);
    bin += __shfl_xor(bin, o, 32);
  }

  if (j == 0) {
    tags[(size_t)b * TT + (TT - 1)] = (unsigned char)bi;
    const float ll = un + bin - logZ;
    atomicAdd(lossp, -ll / (float)BB);
  }
}

// ---------------------------------------------------------------------------
__global__ void backtrace(const unsigned char* __restrict__ bp,
                          unsigned char* __restrict__ tags) {
  if (threadIdx.x != 0) return;
  const int b = blockIdx.x;
  int tag = tags[(size_t)b * TT + (TT - 1)];
  for (int t = TT - 1; t > 0; --t) {
    tag = bp[((size_t)(t - 1) * BB + b) * UU + tag];
    tags[(size_t)b * TT + (t - 1)] = (unsigned char)tag;
  }
}

__global__ void onehot(const unsigned char* __restrict__ tags,
                       float* __restrict__ out) {
  const size_t idx = (size_t)blockIdx.x * blockDim.x + threadIdx.x;
  if (idx >= (size_t)BB * TT * UU) return;
  const size_t bt = idx >> 5;
  const int u = (int)(idx & 31);
  out[idx] = (tags[bt] == (unsigned char)u) ? 1.0f : 0.0f;
}

// ---------------------------------------------------------------------------
extern "C" void kernel_launch(void* const* d_in, const int* in_sizes, int n_in,
                              void* d_out, int out_size, void* d_ws, size_t ws_size,
                              hipStream_t stream) {
  const float* x        = (const float*)d_in[0];
  const int*   y_true   = (const int*)d_in[1];
  const float* base_W   = (const float*)d_in[2];
  const float* base_b   = (const float*)d_in[3];
  const float* crf_K    = (const float*)d_in[4];
  const float* crf_bias = (const float*)d_in[5];
  const float* chain    = (const float*)d_in[6];
  const float* lb       = (const float*)d_in[7];
  const float* rb       = (const float*)d_in[8];
  float* out = (float*)d_out;

  char* ws = (char*)d_ws;
  size_t off = 0;
  float* potbuf = (float*)(ws + off);           off += (size_t)BB * TT * UU * sizeof(float);
  _Float16* Wp  = (_Float16*)(ws + off);        off += (size_t)8 * 16 * 32 * 16 * sizeof(_Float16);
  _Float16* Kp  = (_Float16*)(ws + off);        off += (size_t)8 * 2 * 32 * 16 * sizeof(_Float16);
  unsigned char* bpbuf  = (unsigned char*)(ws + off); off += (size_t)(TT - 1) * BB * UU;
  unsigned char* tagbuf = (unsigned char*)(ws + off); off += (size_t)BB * TT;

  float* lossp = out + (size_t)BB * TT * UU;

  const int packN = 8 * 16 * 32 * 16 + 8 * 2 * 32 * 16;
  pack_weights<<<(packN + 255) / 256, 256, 0, stream>>>(base_W, crf_K, Wp, Kp);
  zero_loss<<<1, 1, 0, stream>>>(lossp);
  pot_gemm<<<(BB * TT) / 64, 128, 0, stream>>>(x, base_b, Wp, Kp, crf_bias, lb, rb, potbuf);
  viterbi_fwd<<<BB, 32, 0, stream>>>(potbuf, chain, y_true, bpbuf, tagbuf, lossp);
  backtrace<<<BB, 1, 0, stream>>>(bpbuf, tagbuf);
  onehot<<<(int)(((size_t)BB * TT * UU + 255) / 256), 256, 0, stream>>>(tagbuf, out);
}
